// GNNEncoder_42872363549026
// MI455X (gfx1250) — matrix operand order repaired and checked by reading, not compiled, not run on hardware
//
#include <hip/hip_runtime.h>
#include <math.h>

// ---------------- problem constants ----------------
#define NN    20000
#define EE    320000
#define HID   64
#define HEADS 4
#define LL    4
#define CC    256          // HID*HEADS
#define HH    512          // JK-LSTM hidden
#define G4    2048         // 4*HH
#define NEG   0.2f
#define EPS   1e-5f

// ---------------- WMMA types ----------------
typedef __bf16 bf16_t;
typedef __attribute__((ext_vector_type(16))) __bf16 v16bf;
typedef __attribute__((ext_vector_type(8)))  float  v8f;

union FragAB {
  v16bf v;
  uint4 q[2];
};

// fp32 -> bf16 (RNE) as raw u16
__device__ __forceinline__ unsigned short f2bf(float f) {
  unsigned int u = __float_as_uint(f);
  u = (u + 0x7FFFu + ((u >> 16) & 1u)) >> 16;
  return (unsigned short)u;
}
__device__ __forceinline__ float sigm(float x) { return 1.0f / (1.0f + __expf(-x)); }

// ordered-uint encoding for float atomicMax (handles negatives)
__device__ __forceinline__ unsigned int encf(float f) {
  unsigned int u = __float_as_uint(f);
  return (u & 0x80000000u) ? ~u : (u | 0x80000000u);
}
__device__ __forceinline__ float decf(unsigned int u) {
  return (u & 0x80000000u) ? __uint_as_float(u & 0x7FFFFFFFu) : __uint_as_float(~u);
}

__device__ __forceinline__ void atomAddF(float* p, float v) { unsafeAtomicAdd(p, v); }

// ---------------- CDNA5 async copy to LDS (ASYNCcnt path) ----------------
// per-lane: LDS[lds_off .. +15] = MEM[g .. +15]
__device__ __forceinline__ void async_ld16(unsigned lds_off, const void* g) {
  asm volatile("global_load_async_to_lds_b128 %0, %1, off"
               :: "v"(lds_off), "v"(g)
               : "memory");
}
__device__ __forceinline__ void wait_async0() {
  asm volatile("s_wait_asynccnt 0x0" ::: "memory");
}

// ======================================================================
// WMMA GEMM:  C[M,Nc] (+)= A[M,K]bf16 * Bt[Nc,K]bf16   (Bt = B column-major)
// block = 256 thr = 8 waves, tile 128(M) x 128(N); wave tile 32x64 (2x4 WMMA)
// A/B tiles staged to LDS via GLOBAL_LOAD_ASYNC_TO_LDS_B128, double-buffered.
// ======================================================================
#define APAD      40                    // padded row stride (bf16 elems) = 80 B
#define TILEBYTES (128 * APAD * 2)      // one staged tile buffer = 10240 B

__global__ __launch_bounds__(256)
void gemm_bf16_wmma(const unsigned short* __restrict__ A,
                    const unsigned short* __restrict__ Bt,
                    float* __restrict__ Cmat,
                    int M, int K, int Nc, int accum)
{
  __shared__ __align__(16) unsigned short Alds[2][128 * APAD];
  __shared__ __align__(16) unsigned short Blds[2][128 * APAD];

  const int tid  = threadIdx.x;
  const int lane = tid & 31;
  const int wave = tid >> 5;
  const int wm   = wave & 3;        // 4 waves along M
  const int wn   = wave >> 2;       // 2 waves along N
  const int half = lane >> 4;
  const int l16  = lane & 15;
  const int m0   = blockIdx.y * 128;
  const int n0   = blockIdx.x * 128;

  // ---- async-copy assignments: 2 A segs + 2 B segs (16 B each) per thread ----
  const unsigned ldsA0 = (unsigned)(uintptr_t)&Alds[0][0];
  const unsigned ldsB0 = (unsigned)(uintptr_t)&Blds[0][0];
  const unsigned short* agp[2];
  const unsigned short* bgp[2];
  unsigned soff[2];
#pragma unroll
  for (int i = 0; i < 2; ++i) {
    int seg = tid + i * 256;          // 0..511 ; covers 128 rows x 4 x 16B
    int r = seg >> 2, s4 = seg & 3;
    int gr = m0 + r; if (gr > M - 1) gr = M - 1;   // clamp; C-store is guarded
    agp[i]  = A  + (size_t)gr * K + s4 * 8;
    bgp[i]  = Bt + (size_t)(n0 + r) * K + s4 * 8;  // Nc is a multiple of 128
    soff[i] = (unsigned)(r * (APAD * 2) + s4 * 16);
  }

  v8f acc[2][4];
  const v8f vz = {0.f,0.f,0.f,0.f,0.f,0.f,0.f,0.f};
#pragma unroll
  for (int i = 0; i < 2; ++i)
#pragma unroll
    for (int j = 0; j < 4; ++j) acc[i][j] = vz;

  // prologue: stage buffer 0 for kk = 0
#pragma unroll
  for (int i = 0; i < 2; ++i) {
    async_ld16(ldsA0 + soff[i], agp[i]);
    async_ld16(ldsB0 + soff[i], bgp[i]);
  }

  int buf = 0;
  for (int kk = 0; kk < K; kk += 32) {
    wait_async0();          // our copies into `buf` are done
    __syncthreads();        // everyone's copies done; everyone done reading buf^1
    if (kk + 32 < K) {
      const int nb = buf ^ 1;
#pragma unroll
      for (int i = 0; i < 2; ++i) {
        async_ld16(ldsA0 + nb * TILEBYTES + soff[i], agp[i] + kk + 32);
        async_ld16(ldsB0 + nb * TILEBYTES + soff[i], bgp[i] + kk + 32);
      }
    }
    FragAB a[2], b[4];
#pragma unroll
    for (int i = 0; i < 2; ++i) {   // A frag: row l16, contiguous 16 K at +16h
      const uint4* p = (const uint4*)&Alds[buf][(wm * 32 + i * 16 + l16) * APAD + 16 * half];
      a[i].q[0] = p[0];
      a[i].q[1] = p[1];
    }
#pragma unroll
    for (int j = 0; j < 4; ++j) {   // B frag: col l16, contiguous 16 K at +16h
      const uint4* p = (const uint4*)&Blds[buf][(wn * 64 + j * 16 + l16) * APAD + 16 * half];
      b[j].q[0] = p[0];
      b[j].q[1] = p[1];
    }
#pragma unroll
    for (int i = 0; i < 2; ++i)
#pragma unroll
      for (int j = 0; j < 4; ++j)
        acc[i][j] = __builtin_amdgcn_wmma_f32_16x16x32_bf16(
            false, a[i].v, false, b[j].v, (short)0, acc[i][j], false, false);
    buf ^= 1;
  }

#pragma unroll
  for (int i = 0; i < 2; ++i)
#pragma unroll
    for (int j = 0; j < 4; ++j) {
      int n = n0 + wn * 64 + j * 16 + l16;
#pragma unroll
      for (int r = 0; r < 8; ++r) {
        int m = m0 + wm * 32 + i * 16 + r + 8 * half;
        if (m < M) {
          float* cp = Cmat + (size_t)m * Nc + n;
          float  v  = acc[i][j][r];
          *cp = accum ? (*cp + v) : v;
        }
      }
    }
}

// ======================================================================
// small utility kernels
// ======================================================================
__global__ __launch_bounds__(256) void fill_f32(float* p, float v, int n) {
  int i = blockIdx.x * 256 + threadIdx.x;
  if (i < n) p[i] = v;
}
__global__ __launch_bounds__(256) void fill_u32(unsigned int* p, unsigned int v, int n) {
  int i = blockIdx.x * 256 + threadIdx.x;
  if (i < n) p[i] = v;
}
__global__ __launch_bounds__(256) void bcast_row(float* p, const float* row, int rowlen_mask, int n) {
  int i = blockIdx.x * 256 + threadIdx.x;
  if (i < n) p[i] = row[i & rowlen_mask];
}
__global__ __launch_bounds__(256) void conv_bf(const float* w, unsigned short* o, int n) {
  int i = blockIdx.x * 256 + threadIdx.x;
  if (i < n) o[i] = f2bf(w[i]);
}
// transpose [K,Nc] fp32 -> [Nc,K] bf16
__global__ __launch_bounds__(256) void conv_transpose_bf(const float* w, unsigned short* o, int K, int Nc) {
  int i = blockIdx.x * 256 + threadIdx.x;
  if (i < K * Nc) {
    int k = i / Nc, n = i % Nc;
    o[(size_t)n * K + k] = f2bf(w[i]);
  }
}

// layer-0 node transforms (K=2, do directly)
__global__ __launch_bounds__(256)
void layer0_xform(const float* __restrict__ x, const float* __restrict__ Ws0,
                  const float* __restrict__ Wd0, float* __restrict__ xs, float* __restrict__ xd) {
  int i = blockIdx.x * 256 + threadIdx.x;
  if (i >= NN * CC) return;
  int n = i >> 8, c = i & 255;
  float x0 = x[n * 2], x1 = x[n * 2 + 1];
  xs[i] = x0 * Ws0[c] + x1 * Ws0[CC + c];
  xd[i] = x0 * Wd0[c] + x1 * Wd0[CC + c];
}

// per-edge attention logits + segment max  (one wave of 32 lanes per edge)
__global__ __launch_bounds__(256)
void edge_logit(const float* __restrict__ xs, const float* __restrict__ xd,
                const float* __restrict__ ea, const float* __restrict__ We,
                const float* __restrict__ attv, const int* __restrict__ srcI,
                const int* __restrict__ dstI, float* __restrict__ logit,
                unsigned int* __restrict__ lmax) {
  int gid  = blockIdx.x * 256 + threadIdx.x;
  int e    = gid >> 5;
  int lane = gid & 31;
  if (e >= EE) return;
  int s = srcI[e], d = dstI[e];
  float av = ea[e];
  const float4* xs4 = (const float4*)(xs + (size_t)s * CC);
  const float4* xd4 = (const float4*)(xd + (size_t)d * CC);
  float partial = 0.f;
#pragma unroll
  for (int q = 0; q < 2; ++q) {
    int c4 = lane * 2 + q;              // float4 index; channels c4*4..+3
    float4 a = xs4[c4], b = xd4[c4];
    int c = c4 * 4;
    float m;
    m = a.x + b.x + av * We[c + 0]; m = m > 0.f ? m : NEG * m; partial += m * attv[c + 0];
    m = a.y + b.y + av * We[c + 1]; m = m > 0.f ? m : NEG * m; partial += m * attv[c + 1];
    m = a.z + b.z + av * We[c + 2]; m = m > 0.f ? m : NEG * m; partial += m * attv[c + 2];
    m = a.w + b.w + av * We[c + 3]; m = m > 0.f ? m : NEG * m; partial += m * attv[c + 3];
  }
  // reduce within the 8 lanes of one head (channels lane*8..+7 -> head = lane/8)
#pragma unroll
  for (int off = 1; off < 8; off <<= 1) partial += __shfl_xor(partial, off, 32);
  if ((lane & 7) == 0) {
    int h = lane >> 3;
    logit[(size_t)e * HEADS + h] = partial;
    atomicMax(&lmax[(size_t)d * HEADS + h], encf(partial));
  }
}

// exp(logit - max) and segment sum
__global__ __launch_bounds__(256)
void edge_exp(const float* __restrict__ logit, const unsigned int* __restrict__ lmax,
              const int* __restrict__ dstI, float* __restrict__ exb, float* __restrict__ den) {
  int i = blockIdx.x * 256 + threadIdx.x;
  if (i >= EE * HEADS) return;
  int e = i >> 2, h = i & 3;
  int d = dstI[e];
  float ex = __expf(logit[i] - decf(lmax[(size_t)d * HEADS + h]));
  exb[i] = ex;
  atomAddF(&den[(size_t)d * HEADS + h], ex);
}

// out[dst] += alpha * xs[src]   (one wave per edge, 8 channels/lane)
__global__ __launch_bounds__(256)
void edge_scatter(const float* __restrict__ xs, const float* __restrict__ exb,
                  const float* __restrict__ den, const int* __restrict__ srcI,
                  const int* __restrict__ dstI, float* __restrict__ out) {
  int gid  = blockIdx.x * 256 + threadIdx.x;
  int e    = gid >> 5;
  int lane = gid & 31;
  if (e >= EE) return;
  int s = srcI[e], d = dstI[e];
  int h = lane >> 3;
  float alpha = exb[(size_t)e * HEADS + h] / (den[(size_t)d * HEADS + h] + 1e-16f);
  const float4* xs4 = (const float4*)(xs + (size_t)s * CC);
  float* o = out + (size_t)d * CC + lane * 8;
#pragma unroll
  for (int q = 0; q < 2; ++q) {
    float4 v = xs4[lane * 2 + q];
    atomAddF(o + q * 4 + 0, v.x * alpha);
    atomAddF(o + q * 4 + 1, v.y * alpha);
    atomAddF(o + q * 4 + 2, v.z * alpha);
    atomAddF(o + q * 4 + 3, v.w * alpha);
  }
}

// ReLU in place + produce bf16 copy (next layer / LSTM input)
__global__ __launch_bounds__(256)
void relu_cvt(float* __restrict__ seqL, unsigned short* __restrict__ seqbfL, int n) {
  int i = blockIdx.x * 256 + threadIdx.x;
  if (i >= n) return;
  float v = seqL[i];
  v = v > 0.f ? v : 0.f;
  seqL[i]   = v;
  seqbfL[i] = f2bf(v);
}

// LSTM pointwise: gates -> c,h ; h->bf16 for next GEMM; score += h . att_w_seg
// one wave per node, 16 channels per lane (strided for coalescing)
__global__ __launch_bounds__(256)
void lstm_update(const float* __restrict__ G, float* __restrict__ cst,
                 unsigned short* __restrict__ hbf, float* __restrict__ score,
                 const float* __restrict__ attw, int lidx) {
  int gid  = blockIdx.x * 256 + threadIdx.x;
  int n    = gid >> 5;
  int lane = gid & 31;
  if (n >= NN) return;
  const float* g = G + (size_t)n * G4;
  float* cs = cst + (size_t)n * HH;
  unsigned short* hb = hbf + (size_t)n * HH;
  float acc = 0.f;
#pragma unroll
  for (int k = 0; k < 16; ++k) {
    int c = lane + 32 * k;
    float ig = sigm(g[c]);
    float fg = sigm(g[HH + c]);
    float gg = tanhf(g[2 * HH + c]);
    float og = sigm(g[3 * HH + c]);
    float cc = fg * cs[c] + ig * gg;
    cs[c] = cc;
    float h = og * tanhf(cc);
    hb[c] = f2bf(h);
    acc += h * attw[c];
  }
#pragma unroll
  for (int off = 1; off < 32; off <<= 1) acc += __shfl_xor(acc, off, 32);
  if (lane == 0) score[(size_t)n * LL + lidx] += acc;
}

// per-node softmax over L scores, jk mix, BN partial sums
__global__ __launch_bounds__(256)
void jk_mix(const float* __restrict__ score, const float* __restrict__ seq,
            float* __restrict__ jk, float* __restrict__ bnsum, float* __restrict__ bnsq) {
  int n = blockIdx.x;
  int c = threadIdx.x;
  float s0 = score[n * LL + 0], s1 = score[n * LL + 1];
  float s2 = score[n * LL + 2], s3 = score[n * LL + 3];
  float m = fmaxf(fmaxf(s0, s1), fmaxf(s2, s3));
  float e0 = __expf(s0 - m), e1 = __expf(s1 - m), e2 = __expf(s2 - m), e3 = __expf(s3 - m);
  float inv = 1.0f / (e0 + e1 + e2 + e3);
  size_t base = (size_t)n * CC + c;
  const size_t stride = (size_t)NN * CC;
  float v = (e0 * seq[base] + e1 * seq[base + stride] +
             e2 * seq[base + 2 * stride] + e3 * seq[base + 3 * stride]) * inv;
  jk[base] = v;
  atomAddF(&bnsum[c], v);
  atomAddF(&bnsq[c], v * v);
}

// BatchNorm (batch stats) + ReLU -> output
__global__ __launch_bounds__(256)
void bn_relu(const float* __restrict__ jk, const float* __restrict__ bnsum,
             const float* __restrict__ bnsq, const float* __restrict__ gamma,
             const float* __restrict__ beta, float* __restrict__ out) {
  int n = blockIdx.x;
  int c = threadIdx.x;
  float mu  = bnsum[c] * (1.0f / NN);
  float var = bnsq[c] * (1.0f / NN) - mu * mu;
  float v = (jk[(size_t)n * CC + c] - mu) * rsqrtf(var + EPS) * gamma[c] + beta[c];
  out[(size_t)n * CC + c] = v > 0.f ? v : 0.f;
}

// ======================================================================
// host-side orchestration
// ======================================================================
static inline int cdiv(long a, long b) { return (int)((a + b - 1) / b); }

extern "C" void kernel_launch(void* const* d_in, const int* in_sizes, int n_in,
                              void* d_out, int out_size, void* d_ws, size_t ws_size,
                              hipStream_t stream) {
  (void)in_sizes; (void)n_in; (void)out_size; (void)ws_size;
  const float* x      = (const float*)d_in[0];
  const int*   ei     = (const int*)d_in[1];
  const float* eattr  = (const float*)d_in[2];
  const float* W_src0 = (const float*)d_in[3];
  const float* W_dst0 = (const float*)d_in[4];
  const float* W_src  = (const float*)d_in[5];
  const float* W_dst  = (const float*)d_in[6];
  const float* W_edge = (const float*)d_in[7];
  const float* att    = (const float*)d_in[8];
  const float* bias   = (const float*)d_in[9];
  const float* Wih_f  = (const float*)d_in[10];
  const float* Whh_f  = (const float*)d_in[11];
  const float* b_f    = (const float*)d_in[12];
  const float* Wih_b  = (const float*)d_in[13];
  const float* Whh_b  = (const float*)d_in[14];
  const float* b_b    = (const float*)d_in[15];
  const float* att_w  = (const float*)d_in[16];
  const float* bn_g   = (const float*)d_in[18];
  const float* bn_b   = (const float*)d_in[19];
  const int* srcI = ei;
  const int* dstI = ei + EE;
  float* out = (float*)d_out;

  // ---- workspace carve-up ----
  char* ws = (char*)d_ws;
  size_t off = 0;
  auto carve = [&](size_t bytes) -> char* {
    char* p = ws + off;
    off = (off + bytes + 255) & ~(size_t)255;
    return p;
  };
  const size_t NC = (size_t)NN * CC;
  float*          seq     = (float*)carve(LL * NC * 4);          // post-act layer outputs
  unsigned short* seq_bf  = (unsigned short*)carve(LL * NC * 2); // bf16 copies
  float*          xs      = (float*)carve(NC * 4);
  float*          xd      = (float*)carve(NC * 4);
  float*          logit   = (float*)carve((size_t)EE * HEADS * 4);
  float*          exb     = (float*)carve((size_t)EE * HEADS * 4);
  unsigned int*   lmax    = (unsigned int*)carve((size_t)NN * HEADS * 4);
  float*          den     = (float*)carve((size_t)NN * HEADS * 4);
  unsigned short* Wsrc_t  = (unsigned short*)carve((size_t)3 * CC * CC * 2);
  unsigned short* Wdst_t  = (unsigned short*)carve((size_t)3 * CC * CC * 2);
  unsigned short* Wihf_bf = (unsigned short*)carve((size_t)G4 * CC * 2);
  unsigned short* Whhf_bf = (unsigned short*)carve((size_t)G4 * HH * 2);
  unsigned short* Wihb_bf = (unsigned short*)carve((size_t)G4 * CC * 2);
  unsigned short* Whhb_bf = (unsigned short*)carve((size_t)G4 * HH * 2);
  float*          G       = (float*)carve((size_t)NN * G4 * 4);
  unsigned short* hbf     = (unsigned short*)carve((size_t)NN * HH * 2);
  float*          cstate  = (float*)carve((size_t)NN * HH * 4);
  float*          score   = (float*)carve((size_t)NN * LL * 4);
  float*          jk      = (float*)carve(NC * 4);
  float*          bnsum   = (float*)carve(CC * 4);
  float*          bnsq    = (float*)carve(CC * 4);

  // ---- weight prep (fp32 -> bf16, GAT weights transposed to [n][k]) ----
  for (int l = 0; l < 3; ++l) {
    conv_transpose_bf<<<cdiv(CC * CC, 256), 256, 0, stream>>>(
        W_src + (size_t)l * CC * CC, Wsrc_t + (size_t)l * CC * CC, CC, CC);
    conv_transpose_bf<<<cdiv(CC * CC, 256), 256, 0, stream>>>(
        W_dst + (size_t)l * CC * CC, Wdst_t + (size_t)l * CC * CC, CC, CC);
  }
  conv_bf<<<cdiv(G4 * CC, 256), 256, 0, stream>>>(Wih_f, Wihf_bf, G4 * CC); // already [n][k]
  conv_bf<<<cdiv(G4 * HH, 256), 256, 0, stream>>>(Whh_f, Whhf_bf, G4 * HH);
  conv_bf<<<cdiv(G4 * CC, 256), 256, 0, stream>>>(Wih_b, Wihb_bf, G4 * CC);
  conv_bf<<<cdiv(G4 * HH, 256), 256, 0, stream>>>(Whh_b, Whhb_bf, G4 * HH);

  // ---- GAT layers ----
  const dim3 gemmB(256);
  for (int l = 0; l < LL; ++l) {
    if (l == 0) {
      layer0_xform<<<cdiv(NC, 256), 256, 0, stream>>>(x, W_src0, W_dst0, xs, xd);
    } else {
      dim3 g(CC / 128, cdiv(NN, 128));
      gemm_bf16_wmma<<<g, gemmB, 0, stream>>>(seq_bf + (size_t)(l - 1) * NC,
                                              Wsrc_t + (size_t)(l - 1) * CC * CC,
                                              xs, NN, CC, CC, 0);
      gemm_bf16_wmma<<<g, gemmB, 0, stream>>>(seq_bf + (size_t)(l - 1) * NC,
                                              Wdst_t + (size_t)(l - 1) * CC * CC,
                                              xd, NN, CC, CC, 0);
    }
    fill_u32<<<cdiv(NN * HEADS, 256), 256, 0, stream>>>(lmax, 0x007FFFFFu /*enc(-inf)*/, NN * HEADS);
    fill_f32<<<cdiv(NN * HEADS, 256), 256, 0, stream>>>(den, 0.f, NN * HEADS);
    bcast_row<<<cdiv(NC, 256), 256, 0, stream>>>(seq + (size_t)l * NC, bias + l * CC, 255, (int)NC);

    edge_logit<<<cdiv((size_t)EE * 32, 256), 256, 0, stream>>>(
        xs, xd, eattr, W_edge + l * CC, att + l * HEADS * HID, srcI, dstI, logit, lmax);
    edge_exp<<<cdiv(EE * HEADS, 256), 256, 0, stream>>>(logit, lmax, dstI, exb, den);
    edge_scatter<<<cdiv((size_t)EE * 32, 256), 256, 0, stream>>>(
        xs, exb, den, srcI, dstI, seq + (size_t)l * NC);
    relu_cvt<<<cdiv(NC, 256), 256, 0, stream>>>(seq + (size_t)l * NC, seq_bf + (size_t)l * NC, (int)NC);
  }

  // ---- JK bi-LSTM over layer axis ----
  fill_f32<<<cdiv(NN * LL, 256), 256, 0, stream>>>(score, 0.f, NN * LL); // att_b cancels in softmax
  for (int dir = 0; dir < 2; ++dir) {
    const unsigned short* Wih = dir == 0 ? Wihf_bf : Wihb_bf;
    const unsigned short* Whh = dir == 0 ? Whhf_bf : Whhb_bf;
    const float* bb           = dir == 0 ? b_f : b_b;
    fill_u32<<<cdiv((size_t)NN * HH / 2, 256), 256, 0, stream>>>((unsigned int*)hbf, 0u, NN * HH / 2);
    fill_f32<<<cdiv((size_t)NN * HH, 256), 256, 0, stream>>>(cstate, 0.f, NN * HH);
    for (int s = 0; s < LL; ++s) {
      int t = dir == 0 ? s : (LL - 1 - s);
      bcast_row<<<cdiv((size_t)NN * G4, 256), 256, 0, stream>>>(G, bb, G4 - 1, NN * G4);
      dim3 g(G4 / 128, cdiv(NN, 128));
      gemm_bf16_wmma<<<g, gemmB, 0, stream>>>(seq_bf + (size_t)t * NC, Wih, G, NN, CC, G4, 1);
      gemm_bf16_wmma<<<g, gemmB, 0, stream>>>(hbf, Whh, G, NN, HH, G4, 1);
      lstm_update<<<cdiv((size_t)NN * 32, 256), 256, 0, stream>>>(
          G, cstate, hbf, score, att_w + dir * HH, t);
    }
  }

  // ---- attention mix + BatchNorm + ReLU ----
  fill_f32<<<1, 256, 0, stream>>>(bnsum, 0.f, CC);
  fill_f32<<<1, 256, 0, stream>>>(bnsq, 0.f, CC);
  jk_mix<<<NN, 256, 0, stream>>>(score, seq, jk, bnsum, bnsq);
  bn_relu<<<NN, 256, 0, stream>>>(jk, bnsum, bnsq, bn_g, bn_b, out);
}
